// GraphProcessor_27977416966194
// MI455X (gfx1250) — matrix operand order repaired and checked by reading, not compiled, not run on hardware
//
#include <hip/hip_runtime.h>
#include <hip/hip_bf16.h>
#include <math.h>

namespace {

constexpr int S_DIM = 16;
constexpr int B_DIM = 64;
constexpr int C_DIM = 128;
constexpr int H_DIM = 11;
constexpr int W_DIM = 11;
constexpr int P_DIM = H_DIM * W_DIM;   // 121
constexpr int NB    = S_DIM * B_DIM;   // 1024
constexpr int CP    = C_DIM * P_DIM;   // 15488 (valid elems per tile)
constexpr int TS    = 128 * 128;       // padded activation tile (T-layout [p][c])
constexpr int T2S   = 128 * 256;       // padded zr tile ([p][2C])
constexpr int ZROW  = 121;             // guaranteed-zero pad row in conv-input tiles
constexpr float LN_EPS   = 1e-5f;
constexpr float QK_SCALE = 0.08838834764831845f;  // 128^-0.5

typedef __bf16 bf16;
typedef __attribute__((ext_vector_type(16))) __bf16 v16bf;
typedef __attribute__((ext_vector_type(8)))  __bf16 v8bf;
typedef __attribute__((ext_vector_type(8)))  float  v8f;

__device__ __forceinline__ int lane_id() { return (int)(threadIdx.x & 31u); }

__device__ __forceinline__ v8f wmma_bf16(v16bf a, v16bf b, v8f c) {
  return __builtin_amdgcn_wmma_f32_16x16x32_bf16(false, a, false, b, (short)0, c, false, false);
}

__device__ __forceinline__ v16bf frag_combine(v8bf lo, v8bf hi) {
  v16bf r;
#pragma unroll
  for (int i = 0; i < 8; ++i) { r[i] = lo[i]; r[8 + i] = hi[i]; }
  return r;
}

// ---- branchless 128-bit fragment loaders (wave32, contiguous-K order shared by A and B) ----
// A 16x32: lane (l,g): row m0+l ; K = k0+16g..+15  (2x b128)
__device__ __forceinline__ v16bf load_a_frag(const bf16* __restrict__ A, int lda,
                                             int m0, int k0) {
  const int l = lane_id() & 15, g = (lane_id() >> 4) & 1;
  const bf16* p = A + (size_t)(m0 + l) * lda + (k0 + 16 * g);
  return frag_combine(*(const v8bf*)p, *(const v8bf*)(p + 8));
}

// B 32x16 from K-contiguous storage: B[k][n] = Src[n*lds + k]; per lane K = k0+16g..+15 (2x b128)
__device__ __forceinline__ v16bf load_b_fragT(const bf16* __restrict__ Src, int lds_,
                                              int k0, int n0) {
  const int l = lane_id() & 15, g = (lane_id() >> 4) & 1;
  const bf16* p = Src + (size_t)(n0 + l) * lds_ + (k0 + 16 * g);
  return frag_combine(*(const v8bf*)p, *(const v8bf*)(p + 8));
}

// B fragment with explicit per-lane row (conv with shifted/zero rows), K-contiguous
__device__ __forceinline__ v16bf load_b_row(const bf16* __restrict__ Src, int lds_,
                                            int k0, int row) {
  const int g = (lane_id() >> 4) & 1;
  const bf16* p = Src + (size_t)row * lds_ + (k0 + 16 * g);
  return frag_combine(*(const v8bf*)p, *(const v8bf*)(p + 8));
}

// ---------------- GEMM kernels ----------------
// out[o][p] = sum_c W[o][c] * act[(s+sShift),b][c][p] + bias   (bias never null)
// act in T-layout [p][c]; optional outputs: OutT (T-layout [p][o]) and OutC ([o][p], pad cols stay 0)
__global__ void k_gemm_w_act(const bf16* __restrict__ W, const bf16* __restrict__ ActT,
                             const float* __restrict__ bias,
                             bf16* __restrict__ OutT, bf16* __restrict__ OutC, int sShift) {
  const int batch = blockIdx.x;
  const int s = batch / B_DIM, b = batch % B_DIM;
  const int ssrc = s + sShift;
  const bool valid = (ssrc >= 0) && (ssrc < S_DIM);
  const int tile = blockIdx.y * 8 + (threadIdx.x >> 5);
  const int mT = (tile >> 3) << 4;
  const int nT = (tile & 7) << 4;
  v8f acc = {};
  if (valid) {
    const bf16* act = ActT + (size_t)(ssrc * B_DIM + b) * TS;
#pragma unroll
    for (int k0 = 0; k0 < C_DIM; k0 += 32) {
      acc = wmma_bf16(load_a_frag(W, C_DIM, mT, k0),
                      load_b_fragT(act, 128, k0, nT), acc);
    }
  }
  const int l = lane_id() & 15, g = (lane_id() >> 4) & 1;
  const int n = nT + l;           // pixel p
  const int mB = mT + 8 * g;      // first of 8 consecutive output channels
  const v8f bvec = *(const v8f*)(bias + mB);   // 32B-aligned vector load
  v8bf ov;
#pragma unroll
  for (int i = 0; i < 8; ++i) ov[i] = (bf16)(acc[i] + bvec[i]);
  if (OutT) *(v8bf*)(OutT + (size_t)batch * TS + n * 128 + mB) = ov;  // pad rows unread
  if (OutC && n < P_DIM) {        // pad cols must stay zero (K-pad of att*V GEMM)
    bf16* oc = OutC + (size_t)batch * TS + n;
#pragma unroll
    for (int i = 0; i < 8; ++i) oc[(mB + i) * 128] = ov[i];
  }
}

// L[p][q] = scale * sum_c Qa_T[p][c] * Ka_T[q][c]   (padded 128x128 f32; pads unread)
__global__ void k_gemm_qk(const bf16* __restrict__ QaT, const bf16* __restrict__ KaT,
                          float* __restrict__ L, float scale) {
  const int batch = blockIdx.x;
  const int tile = blockIdx.y * 8 + (threadIdx.x >> 5);
  const int mT = (tile >> 3) << 4;
  const int nT = (tile & 7) << 4;
  const bf16* q = QaT + (size_t)batch * TS;
  const bf16* k = KaT + (size_t)batch * TS;
  v8f acc = {};
#pragma unroll
  for (int k0 = 0; k0 < C_DIM; k0 += 32) {
    acc = wmma_bf16(load_a_frag(q, 128, mT, k0), load_b_fragT(k, 128, k0, nT), acc);
  }
  const int l = lane_id() & 15, g = (lane_id() >> 4) & 1;
  const int n = nT + l;
  float* out = L + (size_t)batch * TS;
#pragma unroll
  for (int i = 0; i < 8; ++i) out[(mT + i + 8 * g) * 128 + n] = acc[i] * scale;
}

// X_T[p][c] = sum_q Att[p][q] * Vc[c][q]   (pad rows of X_T get garbage; unread)
__global__ void k_gemm_av(const bf16* __restrict__ Att, const bf16* __restrict__ Vc,
                          float* __restrict__ XT) {
  const int batch = blockIdx.x;
  const int tile = blockIdx.y * 8 + (threadIdx.x >> 5);
  const int mT = (tile >> 3) << 4;   // p
  const int nT = (tile & 7) << 4;    // c
  const bf16* att = Att + (size_t)batch * TS;
  const bf16* v   = Vc  + (size_t)batch * TS;
  v8f acc = {};
#pragma unroll
  for (int k0 = 0; k0 < 128; k0 += 32) {   // K = q padded; att cols / v cols >=121 are zero
    acc = wmma_bf16(load_a_frag(att, 128, mT, k0), load_b_fragT(v, 128, k0, nT), acc);
  }
  const int l = lane_id() & 15, g = (lane_id() >> 4) & 1;
  const int n = nT + l;
  float* out = XT + (size_t)batch * TS;
#pragma unroll
  for (int i = 0; i < 8; ++i) out[(mT + i + 8 * g) * 128 + n] = acc[i];
}

// ---------------- 3x3 convs (implicit GEMM, branchless zero-row padding) ----------------
// ZrT[p][o] = sigmoid(conv3(concat[In0,In1])[o][p] + bzr[o]),  o in [0,256)
__global__ void k_conv_zr(const bf16* __restrict__ Wt, const float* __restrict__ bzr,
                          const bf16* __restrict__ In0T, const bf16* __restrict__ In1T,
                          float* __restrict__ ZrT) {
  const int batch = blockIdx.x;
  const int tile = blockIdx.y * 8 + (threadIdx.x >> 5);   // 128 tiles: 16 m x 8 n
  const int mT = (tile >> 3) << 4;
  const int nT = (tile & 7) << 4;
  const bf16* in0 = In0T + (size_t)batch * TS;
  const bf16* in1 = In1T + (size_t)batch * TS;
  const int l = lane_id() & 15, g = (lane_id() >> 4) & 1;
  const int n = nT + l;
  const int py = n / W_DIM, px = n % W_DIM;
  v8f acc = {};
#pragma unroll 1
  for (int tap = 0; tap < 9; ++tap) {
    const int dy = tap / 3 - 1, dx = tap % 3 - 1;
    const int sy = py + dy, sx = px + dx;
    const bool inb = (n < P_DIM) && (sy >= 0) && (sy < H_DIM) && (sx >= 0) && (sx < W_DIM);
    const int row = inb ? (sy * W_DIM + sx) : ZROW;   // ZROW row is all-zero
    const bf16* wt = Wt + (size_t)tap * 256 * 256;
#pragma unroll
    for (int k0 = 0; k0 < 256; k0 += 32) {
      const bf16* src = (k0 < C_DIM) ? in0 : in1;
      acc = wmma_bf16(load_a_frag(wt, 256, mT, k0),
                      load_b_row(src, 128, k0 & 127, row), acc);
    }
  }
  const int mB = mT + 8 * g;
  const v8f bvec = *(const v8f*)(bzr + mB);
  v8f o;
#pragma unroll
  for (int i = 0; i < 8; ++i) o[i] = 1.0f / (1.0f + __expf(-(acc[i] + bvec[i])));
  *(v8f*)(ZrT + (size_t)batch * T2S + n * 256 + mB) = o;   // pad rows unread
}

// h_hat conv + fused GRU: NhT[p][c] = (1-z)*h + z*tanh(conv+b) + h
__global__ void k_conv_h(const bf16* __restrict__ Wt, const float* __restrict__ bh,
                         const bf16* __restrict__ In0T, const bf16* __restrict__ In1T,
                         const float* __restrict__ ZrT, const float* __restrict__ HfT,
                         float* __restrict__ NhT) {
  const int batch = blockIdx.x;
  const int tile = blockIdx.y * 8 + (threadIdx.x >> 5);   // 64 tiles
  const int mT = (tile >> 3) << 4;
  const int nT = (tile & 7) << 4;
  const bf16* in0 = In0T + (size_t)batch * TS;
  const bf16* in1 = In1T + (size_t)batch * TS;
  const int l = lane_id() & 15, g = (lane_id() >> 4) & 1;
  const int n = nT + l;
  const int py = n / W_DIM, px = n % W_DIM;
  v8f acc = {};
#pragma unroll 1
  for (int tap = 0; tap < 9; ++tap) {
    const int dy = tap / 3 - 1, dx = tap % 3 - 1;
    const int sy = py + dy, sx = px + dx;
    const bool inb = (n < P_DIM) && (sy >= 0) && (sy < H_DIM) && (sx >= 0) && (sx < W_DIM);
    const int row = inb ? (sy * W_DIM + sx) : ZROW;
    const bf16* wt = Wt + (size_t)tap * C_DIM * 256;
#pragma unroll
    for (int k0 = 0; k0 < 256; k0 += 32) {
      const bf16* src = (k0 < C_DIM) ? in0 : in1;
      acc = wmma_bf16(load_a_frag(wt, 256, mT, k0),
                      load_b_row(src, 128, k0 & 127, row), acc);
    }
  }
  const int mB = mT + 8 * g;
  const v8f bvec = *(const v8f*)(bh + mB);
  const v8f zv   = *(const v8f*)(ZrT + (size_t)batch * T2S + n * 256 + mB);
  const v8f hv   = *(const v8f*)(HfT + (size_t)batch * TS + n * 128 + mB);
  v8f o;
#pragma unroll
  for (int i = 0; i < 8; ++i) {
    const float hh = tanhf(acc[i] + bvec[i]);
    o[i] = (1.0f - zv[i]) * hv[i] + zv[i] * hh + hv[i];
  }
  *(v8f*)(NhT + (size_t)batch * TS + n * 128 + mB) = o;    // pad rows unread
}

// ---------------- reductions / pointwise ----------------
__global__ void k_softmax(const float* __restrict__ L, bf16* __restrict__ Att) {
  const int row = blockIdx.x;             // NB * 121
  const int batch = row / P_DIM, p = row % P_DIM;
  const float* src = L + (size_t)batch * TS + p * 128;
  const int t = threadIdx.x;              // 128
  float v = (t < P_DIM) ? src[t] : -3.0e38f;
  __shared__ float sh[4];
  float m = v;
#pragma unroll
  for (int o = 16; o > 0; o >>= 1) m = fmaxf(m, __shfl_xor(m, o, 32));
  if ((t & 31) == 0) sh[t >> 5] = m;
  __syncthreads();
  m = fmaxf(fmaxf(sh[0], sh[1]), fmaxf(sh[2], sh[3]));
  __syncthreads();
  const float e = (t < P_DIM) ? __expf(v - m) : 0.0f;
  float ssum = e;
#pragma unroll
  for (int o = 16; o > 0; o >>= 1) ssum += __shfl_xor(ssum, o, 32);
  if ((t & 31) == 0) sh[t >> 5] = ssum;
  __syncthreads();
  ssum = sh[0] + sh[1] + sh[2] + sh[3];
  Att[(size_t)batch * TS + p * 128 + t] = (t < P_DIM) ? (bf16)(e / ssum) : (bf16)0.0f;
}

__global__ void k_meanp(const float* __restrict__ XT, float* __restrict__ Mbar) {
  const int rc = blockIdx.x;              // NB * C
  const int batch = rc >> 7, c = rc & 127;
  const int t = threadIdx.x;              // 128
  float v = (t < P_DIM) ? XT[(size_t)batch * TS + t * 128 + c] : 0.0f;
  __shared__ float sh[4];
#pragma unroll
  for (int o = 16; o > 0; o >>= 1) v += __shfl_xor(v, o, 32);
  if ((t & 31) == 0) sh[t >> 5] = v;
  __syncthreads();
  if (t == 0) Mbar[rc] = (sh[0] + sh[1] + sh[2] + sh[3]) / (float)P_DIM;
}

__global__ void k_gate(const float* __restrict__ Wg, const float* __restrict__ bg,
                       const float* __restrict__ Mbar, float* __restrict__ Gate) {
  const int batch = blockIdx.x;
  const int d = threadIdx.x;  // 128
  const float* mb = Mbar + (size_t)batch * C_DIM;
  float s = bg[d];
  for (int c = 0; c < C_DIM; ++c) s += Wg[d * C_DIM + c] * mb[c];
  Gate[(size_t)batch * C_DIM + d] = 1.0f / (1.0f + __expf(-s));
}

__global__ void k_layernorm(const float* __restrict__ NhT, const float* __restrict__ gamma,
                            const float* __restrict__ beta, float* __restrict__ HfT,
                            bf16* __restrict__ HbT) {
  const int batch = blockIdx.x;
  const float* src = NhT + (size_t)batch * TS;   // valid region = first CP elems
  float s = 0.0f, s2 = 0.0f;
  for (int i = threadIdx.x; i < CP; i += 256) { const float v = src[i]; s += v; s2 += v * v; }
  __shared__ float sh1[8], sh2[8];
#pragma unroll
  for (int o = 16; o > 0; o >>= 1) { s += __shfl_xor(s, o, 32); s2 += __shfl_xor(s2, o, 32); }
  if ((threadIdx.x & 31) == 0) { sh1[threadIdx.x >> 5] = s; sh2[threadIdx.x >> 5] = s2; }
  __syncthreads();
  float ts = 0.0f, ts2 = 0.0f;
#pragma unroll
  for (int w = 0; w < 8; ++w) { ts += sh1[w]; ts2 += sh2[w]; }
  const float mu  = ts / (float)CP;
  const float var = ts2 / (float)CP - mu * mu;
  const float inv = rsqrtf(var + LN_EPS);
  float* hf = HfT + (size_t)batch * TS;
  bf16*  hb = HbT + (size_t)batch * TS;
  for (int i = threadIdx.x; i < CP; i += 256) {
    const int c = i & 127, p = i >> 7;
    const float v = (src[i] - mu) * inv * gamma[c * P_DIM + p] + beta[c * P_DIM + p];
    hf[i] = v;
    hb[i] = (bf16)v;
  }
}

__global__ void k_init(const float* __restrict__ x, float* __restrict__ HfT,
                       bf16* __restrict__ HbT, int n) {
  const int i = blockIdx.x * 256 + threadIdx.x;
  if (i >= n) return;
  const int batch = i / CP, rem = i % CP;
  const int c = rem / P_DIM, p = rem % P_DIM;
  const size_t a = (size_t)batch * TS + p * 128 + c;
  const float v = x[i];
  HfT[a] = v;
  HbT[a] = (bf16)v;
}

__global__ void k_out(const float* __restrict__ HfT, float* __restrict__ out, int n) {
  const int i = blockIdx.x * 256 + threadIdx.x;
  if (i >= n) return;
  const int batch = i / CP, rem = i % CP;
  const int c = rem / P_DIM, p = rem % P_DIM;
  out[i] = HfT[(size_t)batch * TS + p * 128 + c];
}

__global__ void k_cast_valid(const float* __restrict__ in, bf16* __restrict__ out, int n) {
  const int i = blockIdx.x * 256 + threadIdx.x;
  if (i >= n) return;
  const size_t a = (size_t)(i / CP) * TS + (i % CP);
  out[a] = (bf16)in[a];
}

__global__ void k_combine(const float* __restrict__ XT, const float* __restrict__ HfT,
                          const float* __restrict__ alpha, const float* __restrict__ weight,
                          float* __restrict__ CombT, int n) {
  const int i = blockIdx.x * 256 + threadIdx.x;
  if (i >= n) return;
  const size_t a = (size_t)(i / CP) * TS + (i % CP);
  CombT[a] = weight[0] * (alpha[0] * XT[a] + HfT[a]);
}

__global__ void k_addinter(const float* __restrict__ XT, const float* __restrict__ Gate,
                           const float* __restrict__ weight, int dir,
                           float* __restrict__ CombT, int n) {
  const int i = blockIdx.x * 256 + threadIdx.x;
  if (i >= n) return;
  const int batch = i / CP, rem = i % CP;
  const int c = rem & 127;
  const int s = batch / B_DIM;
  const float mask = (dir == 0) ? (s >= 1 ? 1.0f : 0.0f) : (s <= S_DIM - 2 ? 1.0f : 0.0f);
  const size_t a = (size_t)batch * TS + rem;
  CombT[a] += (1.0f - weight[0]) * mask * XT[a] * Gate[(size_t)batch * C_DIM + c];
}

__global__ void k_make_rh(const float* __restrict__ ZrT, const float* __restrict__ HfT,
                          bf16* __restrict__ RhT, int n) {
  const int i = blockIdx.x * 256 + threadIdx.x;
  if (i >= n) return;
  const int batch = i / CP, rem = i % CP;
  const int c = rem & 127, p = rem >> 7;
  const float r = ZrT[(size_t)batch * T2S + p * 256 + 128 + c];
  const size_t a = (size_t)batch * TS + rem;
  RhT[a] = (bf16)(r * HfT[a]);
}

__global__ void k_zero32(unsigned int* __restrict__ p, int n) {
  const int i = blockIdx.x * 256 + threadIdx.x;
  if (i < n) p[i] = 0u;
}

__global__ void k_cast_f32bf(const float* __restrict__ in, bf16* __restrict__ out, int n) {
  const int i = blockIdx.x * 256 + threadIdx.x;
  if (i < n) out[i] = (bf16)in[i];
}

__global__ void k_cast_wmsg(const float* __restrict__ wmsg, bf16* __restrict__ out, int n) {
  const int i = blockIdx.x * 256 + threadIdx.x;
  if (i >= n) return;
  const int o = i / C_DIM, c = i % C_DIM;
  out[i] = (bf16)wmsg[o * (C_DIM + 1) + c];
}

__global__ void k_bias_eff(const float* __restrict__ wmsg, const float* __restrict__ bmsg,
                           float* __restrict__ beL, float* __restrict__ beR) {
  const int d = threadIdx.x;  // 128
  const float edge = wmsg[d * (C_DIM + 1) + C_DIM];
  beL[d] = bmsg[d] + edge;   // edge_val = +1 (left)
  beR[d] = bmsg[d] - edge;   // edge_val = -1 (right)
}

// [O][I][3][3] f32 -> [tap][O][I] bf16
__global__ void k_cast_convw(const float* __restrict__ Win, bf16* __restrict__ Wout,
                             int O, int I) {
  const int idx = blockIdx.x * 256 + threadIdx.x;
  const int tot = O * I * 9;
  if (idx >= tot) return;
  const int tap = idx / (O * I);
  const int rem = idx - tap * (O * I);
  const int o = rem / I, ii = rem % I;
  Wout[idx] = (bf16)Win[(size_t)(o * I + ii) * 9 + tap];
}

}  // namespace

extern "C" void kernel_launch(void* const* d_in, const int* in_sizes, int n_in,
                              void* d_out, int out_size, void* d_ws, size_t ws_size,
                              hipStream_t stream) {
  (void)in_sizes; (void)n_in; (void)out_size; (void)ws_size;
  const float* x      = (const float*)d_in[0];
  const float* wq     = (const float*)d_in[1];
  const float* bq     = (const float*)d_in[2];
  const float* wk     = (const float*)d_in[3];
  const float* bk     = (const float*)d_in[4];
  const float* wv     = (const float*)d_in[5];
  const float* bv     = (const float*)d_in[6];
  const float* alpha  = (const float*)d_in[7];
  const float* wmsg   = (const float*)d_in[8];
  const float* bmsg   = (const float*)d_in[9];
  const float* wlin   = (const float*)d_in[10];
  const float* weight = (const float*)d_in[11];
  const float* wgate  = (const float*)d_in[12];
  const float* bgate  = (const float*)d_in[13];
  const float* wzr    = (const float*)d_in[14];
  const float* bzr    = (const float*)d_in[15];
  const float* wh     = (const float*)d_in[16];
  const float* bh     = (const float*)d_in[17];
  const float* gamma  = (const float*)d_in[18];
  const float* beta   = (const float*)d_in[19];

  char* wsp = (char*)d_ws;
  auto carve = [&](size_t bytes) {
    char* p = wsp;
    wsp += (bytes + 255) & ~(size_t)255;
    return p;
  };

  const size_t TSB_F = (size_t)NB * TS * 4;   // padded f32 act
  const size_t TSB_H = (size_t)NB * TS * 2;   // padded bf16 act

  float* h_f32   = (float*)carve(TSB_F);
  float* logits  = (float*)carve(TSB_F);
  float* comb    = (float*)carve(TSB_F);
  float* Xbuf    = (float*)carve(TSB_F);              // intra / m / nh
  float* zrbuf   = (float*)carve((size_t)NB * T2S * 4);
  bf16*  h_bf    = (bf16*) carve(TSB_H);              // T-layout
  bf16*  q_bf    = (bf16*) carve(TSB_H);              // T ; reused as t
  bf16*  kk_bf   = (bf16*) carve(TSB_H);              // T ; reused as yy_T
  bf16*  v_bf    = (bf16*) carve(TSB_H);              // [c][p] ; reused as yy_C
  bf16*  att_bf  = (bf16*) carve(TSB_H);
  bf16*  comb_bf = (bf16*) carve(TSB_H);              // T
  bf16*  rh_bf   = (bf16*) carve(TSB_H);              // T
  float* mbar    = (float*)carve((size_t)NB * C_DIM * 4);
  float* gate    = (float*)carve((size_t)NB * C_DIM * 4);
  bf16*  wq_bf   = (bf16*) carve((size_t)C_DIM * C_DIM * 2);
  bf16*  wk_bf   = (bf16*) carve((size_t)C_DIM * C_DIM * 2);
  bf16*  wv_bf   = (bf16*) carve((size_t)C_DIM * C_DIM * 2);
  bf16*  wlin_bf = (bf16*) carve((size_t)C_DIM * C_DIM * 2);
  bf16*  wmsg_bf = (bf16*) carve((size_t)C_DIM * C_DIM * 2);
  bf16*  wzr_bf  = (bf16*) carve((size_t)9 * 256 * 256 * 2);
  bf16*  wh_bf   = (bf16*) carve((size_t)9 * 128 * 256 * 2);
  float* beL     = (float*)carve((size_t)C_DIM * 4);
  float* beR     = (float*)carve((size_t)C_DIM * 4);
  float* bzero   = (float*)carve((size_t)C_DIM * 4);  // zero bias for w_lin GEMM

  const int  NTOT = NB * CP;                 // valid elements
  const int  PW   = (NTOT + 255) / 256;
  const dim3 BLK(256);
  const dim3 GEMM_GRID(NB, 8);               // 64 tiles, 8 waves/block
  const dim3 CONV_ZR_GRID(NB, 16);           // 128 tiles (Cout=256)
  const int  nmat = C_DIM * C_DIM;

  // ---- zero bf16 activation buffers (pad rows/cols must be exact zeros) ----
  {
    const int n32 = (int)(TSB_H / 4);
    const int zb  = (n32 + 255) / 256;
    k_zero32<<<zb, BLK, 0, stream>>>((unsigned int*)h_bf, n32);
    k_zero32<<<zb, BLK, 0, stream>>>((unsigned int*)q_bf, n32);
    k_zero32<<<zb, BLK, 0, stream>>>((unsigned int*)kk_bf, n32);
    k_zero32<<<zb, BLK, 0, stream>>>((unsigned int*)v_bf, n32);
    k_zero32<<<zb, BLK, 0, stream>>>((unsigned int*)att_bf, n32);
    k_zero32<<<zb, BLK, 0, stream>>>((unsigned int*)comb_bf, n32);
    k_zero32<<<zb, BLK, 0, stream>>>((unsigned int*)rh_bf, n32);
    k_zero32<<<1, BLK, 0, stream>>>((unsigned int*)bzero, C_DIM);
  }

  // ---- weight prep (bf16) ----
  k_cast_f32bf<<<(nmat + 255) / 256, BLK, 0, stream>>>(wq, wq_bf, nmat);
  k_cast_f32bf<<<(nmat + 255) / 256, BLK, 0, stream>>>(wk, wk_bf, nmat);
  k_cast_f32bf<<<(nmat + 255) / 256, BLK, 0, stream>>>(wv, wv_bf, nmat);
  k_cast_f32bf<<<(nmat + 255) / 256, BLK, 0, stream>>>(wlin, wlin_bf, nmat);
  k_cast_wmsg <<<(nmat + 255) / 256, BLK, 0, stream>>>(wmsg, wmsg_bf, nmat);
  k_bias_eff  <<<1, 128, 0, stream>>>(wmsg, bmsg, beL, beR);
  k_cast_convw<<<(9 * 256 * 256 + 255) / 256, BLK, 0, stream>>>(wzr, wzr_bf, 256, 256);
  k_cast_convw<<<(9 * 128 * 256 + 255) / 256, BLK, 0, stream>>>(wh, wh_bf, 128, 256);

  k_init<<<PW, BLK, 0, stream>>>(x, h_f32, h_bf, NTOT);

  for (int it = 0; it < 3; ++it) {
    // intra attention
    k_gemm_w_act<<<GEMM_GRID, BLK, 0, stream>>>(wq_bf, h_bf, bq, q_bf, nullptr, 0);
    k_gemm_w_act<<<GEMM_GRID, BLK, 0, stream>>>(wk_bf, h_bf, bk, kk_bf, nullptr, 0);
    k_gemm_w_act<<<GEMM_GRID, BLK, 0, stream>>>(wv_bf, h_bf, bv, nullptr, v_bf, 0);
    k_gemm_qk<<<GEMM_GRID, BLK, 0, stream>>>(q_bf, kk_bf, logits, QK_SCALE);
    k_softmax<<<NB * P_DIM, 128, 0, stream>>>(logits, att_bf);
    k_gemm_av<<<GEMM_GRID, BLK, 0, stream>>>(att_bf, v_bf, Xbuf);
    k_combine<<<PW, BLK, 0, stream>>>(Xbuf, h_f32, alpha, weight, comb, NTOT);

    // inter (gated messages, left then right)
    for (int dir = 0; dir < 2; ++dir) {
      const int sShift = (dir == 0) ? -1 : 1;
      const float* be = (dir == 0) ? beL : beR;
      k_gemm_w_act<<<GEMM_GRID, BLK, 0, stream>>>(wmsg_bf, h_bf, be, q_bf, nullptr, sShift); // t (T)
      k_gemm_w_act<<<GEMM_GRID, BLK, 0, stream>>>(wlin_bf, q_bf, bzero, kk_bf, v_bf, 0);     // yy (T + C)
      k_gemm_qk<<<GEMM_GRID, BLK, 0, stream>>>(h_bf, kk_bf, logits, 1.0f);
      k_softmax<<<NB * P_DIM, 128, 0, stream>>>(logits, att_bf);
      k_gemm_av<<<GEMM_GRID, BLK, 0, stream>>>(att_bf, v_bf, Xbuf);                          // m
      k_meanp<<<NB * C_DIM, 128, 0, stream>>>(Xbuf, mbar);
      k_gate<<<NB, 128, 0, stream>>>(wgate, bgate, mbar, gate);
      k_addinter<<<PW, BLK, 0, stream>>>(Xbuf, gate, weight, dir, comb, NTOT);
    }

    // GRU (implicit-GEMM convs) + LayerNorm
    k_cast_valid<<<PW, BLK, 0, stream>>>(comb, comb_bf, NTOT);
    k_conv_zr<<<CONV_ZR_GRID, BLK, 0, stream>>>(wzr_bf, bzr, comb_bf, h_bf, zrbuf);
    k_make_rh<<<PW, BLK, 0, stream>>>(zrbuf, h_f32, rh_bf, NTOT);
    k_conv_h<<<GEMM_GRID, BLK, 0, stream>>>(wh_bf, bh, comb_bf, rh_bf, zrbuf, h_f32, Xbuf);
    k_layernorm<<<NB, 256, 0, stream>>>(Xbuf, gamma, beta, h_f32, h_bf);
  }

  k_out<<<PW, BLK, 0, stream>>>(h_f32, (float*)d_out, NTOT);
}
